// MultilayerGRU_57432302682205
// MI455X (gfx1250) — matrix operand order repaired and checked by reading, not compiled, not run on hardware
//
#include <hip/hip_runtime.h>
#include <hip/hip_bf16.h>

#define Bdim 16
#define Sdim 2048
#define Idim 128
#define Hdim 512
#define Odim 128
#define Ldim 2

typedef __bf16 bf16;
typedef __attribute__((ext_vector_type(16))) __bf16 v16bf;
typedef __attribute__((ext_vector_type(8)))  float  v8f;
typedef __attribute__((ext_vector_type(4)))  int    v4i;

// ---------------------------------------------------------------------------
// WMMA helpers (V_WMMA_F32_16X16X32_BF16, wave32, D = A(16x32) * B(32x16) + C)
// ---------------------------------------------------------------------------
__device__ __forceinline__ v8f wmma_bf16f32(v16bf a, v16bf b, v8f c) {
  // args: (neg_a, A, neg_b, B, c_mod, C, reuse_a, reuse_b)
  return __builtin_amdgcn_wmma_f32_16x16x32_bf16(false, a, false, b, (short)0, c,
                                                 false, false);
}

// A fragment: 16x32 bf16 tile from row-major source (row r at base + r*stride).
// ISA layout: lanes 0-15 -> M=lane, K {0..7,16..23}; lanes 16-31 -> K {8..15,24..31}.
__device__ __forceinline__ v16bf load_a_frag(const bf16* base, int row_stride, int lane) {
  const bf16* p = base + (size_t)(lane & 15) * row_stride + ((lane >> 4) << 3);
  v16bf a;
#pragma unroll
  for (int j = 0; j < 8; ++j) { a[j] = p[j]; a[j + 8] = p[j + 16]; }
  return a;
}

// B fragment: 32x16 (KxN) with B[k][n] = W[n0+n][k0+k], W row-major [Ntot, K].
// ISA layout: lanes 0-15 -> N=lane, K {0..7,16..23}; lanes 16-31 -> N=lane-16, K {8..15,24..31}.
__device__ __forceinline__ v16bf load_b_frag(const bf16* W, int K, int n0, int k0, int lane) {
  const bf16* p = W + (size_t)(n0 + (lane & 15)) * K + k0 + ((lane >> 4) << 3);
  v16bf b;
#pragma unroll
  for (int j = 0; j < 8; ++j) { b[j] = p[j]; b[j + 8] = p[j + 16]; }
  return b;
}

// ---------------------------------------------------------------------------
// Async copy global -> LDS (16 bytes/lane/instruction), tracked by ASYNCcnt.
// Builtin signature (probe-confirmed by round-2 diagnostic):
//   (v4i addrspace(1)* gsrc, v4i addrspace(3)* ldst, imm offset, imm cpol)
// Pointers are produced via integer casts (inttoptr) to sidestep AS-cast rules;
// for LDS the flat address's low 32 bits are the LDS byte offset.
// ---------------------------------------------------------------------------
typedef __attribute__((address_space(1))) v4i as1_v4i;
typedef __attribute__((address_space(3))) v4i as3_v4i;

__device__ __forceinline__ void async_load_b128(const float* gsrc, float* ldst) {
#if __has_builtin(__builtin_amdgcn_global_load_async_to_lds_b128)
  __builtin_amdgcn_global_load_async_to_lds_b128(
      (as1_v4i*)(size_t)gsrc,
      (as3_v4i*)(unsigned)(size_t)ldst,
      0, 0);
#else
  unsigned loff = (unsigned)(size_t)ldst;
  asm volatile("global_load_async_to_lds_b128 %0, %1, off"
               :: "v"(loff), "v"(gsrc) : "memory");
#endif
}

__device__ __forceinline__ void wait_async0() {
#if __has_builtin(__builtin_amdgcn_s_wait_asynccnt)
  __builtin_amdgcn_s_wait_asynccnt(0);
#else
  asm volatile("s_wait_asynccnt 0x0" ::: "memory");
#endif
}

// Prefetch one timestep's xz/xr slice (2 x 16x512 fp32 = 64 KB) into LDS.
// 1024 threads x 16 floats each; waves 0-15 cover xz, waves 16-31 cover xr,
// so the wave-level branch is uniform (4 async issues per wave either way).
__device__ __forceinline__ void async_fetch_step(const float* __restrict__ xz,
                                                 const float* __restrict__ xr,
                                                 float* dz, float* dr,
                                                 int tid, int t) {
  const int o = tid * 16;
  const float* gsrc;
  float* ldst;
  if (o < Bdim * Hdim) {
    gsrc = xz + (size_t)t * (Bdim * Hdim) + o;
    ldst = dz + o;
  } else {
    gsrc = xr + (size_t)t * (Bdim * Hdim) + (o - Bdim * Hdim);
    ldst = dr + (o - Bdim * Hdim);
  }
#pragma unroll
  for (int j = 0; j < 16; j += 4) async_load_b128(gsrc + j, ldst + j);
}

// ---------------------------------------------------------------------------
// fp32 -> bf16 elementwise convert
// ---------------------------------------------------------------------------
__global__ void cvt_f32_bf16(const float* __restrict__ in, bf16* __restrict__ out, int n) {
  int i = blockIdx.x * blockDim.x + threadIdx.x;
  if (i < n) out[i] = (bf16)in[i];
}

// ---------------------------------------------------------------------------
// Input projections: xz[t][b][h] = bias_z[h] + sum_k A[b][t][k] * Wz[h][k]
// Block = 2 waves (wave0: z, wave1: r), grid = (S, H/16).
// ---------------------------------------------------------------------------
__global__ void xproj_kernel(const bf16* __restrict__ A, long a_t_stride, long a_row_stride,
                             int K,
                             const bf16* __restrict__ Wz, const bf16* __restrict__ Wr,
                             const float* __restrict__ bz, const float* __restrict__ br,
                             float* __restrict__ xz, float* __restrict__ xr) {
  const int t    = blockIdx.x;
  const int n0   = blockIdx.y * 16;
  const int wave = threadIdx.x >> 5;
  const int lane = threadIdx.x & 31;

  const bf16*  W    = wave ? Wr : Wz;
  const float* bias = wave ? br : bz;
  float*       out  = wave ? xr : xz;

  const bf16* a_base = A + (size_t)t * a_t_stride;
  v8f acc = {};
  for (int k = 0; k < K; k += 32) {
    v16bf a = load_a_frag(a_base + k, (int)a_row_stride, lane);
    v16bf b = load_b_frag(W, K, n0, k, lane);
    acc = wmma_bf16f32(a, b, acc);
  }

  const int   col   = n0 + (lane & 15);
  const int   mbase = (lane >> 4) << 3;
  const float bb    = bias[col];
  float* o = out + (size_t)t * (Bdim * Hdim) + col;
#pragma unroll
  for (int i = 0; i < 8; ++i) o[(size_t)(mbase + i) * Hdim] = acc[i] + bb;
}

// ---------------------------------------------------------------------------
// GRU recurrence for one layer. Single workgroup of 32 waves; wave w owns
// hidden columns [16w, 16w+16). fp32 master h lives in registers (8/lane);
// bf16 h copy in LDS feeds the shared A-fragments. xz/xr slices are
// double-buffered in LDS via async global->LDS copies overlapping the GEMMs.
// ---------------------------------------------------------------------------
__global__ void __launch_bounds__(1024, 1)
gru_scan_kernel(const float* __restrict__ xz, const float* __restrict__ xr,
                const bf16* __restrict__ Whz, const bf16* __restrict__ Whr,
                const float* __restrict__ h0, int h0_stride,
                bf16* __restrict__ y, float* __restrict__ hT, int hT_stride) {
  __shared__ bf16  h_bf [Bdim][Hdim];            // 16 KB
  __shared__ bf16  rh_bf[Bdim][Hdim];            // 16 KB
  __shared__ float xz_lds[2][Bdim * Hdim];       // 64 KB
  __shared__ float xr_lds[2][Bdim * Hdim];       // 64 KB

  const int tid   = threadIdx.x;
  const int wave  = tid >> 5;
  const int lane  = tid & 31;
  const int n0    = wave * 16;
  const int col   = n0 + (lane & 15);
  const int mbase = (lane >> 4) << 3;

  // fp32 hidden state for this lane's (m, col) slots, resident across the scan
  float hreg[8];
#pragma unroll
  for (int i = 0; i < 8; ++i)
    hreg[i] = h0[(size_t)(mbase + i) * h0_stride + col];

  for (int idx = tid; idx < Bdim * Hdim; idx += 1024) {
    int b = idx >> 9, hh = idx & (Hdim - 1);
    h_bf[b][hh] = (bf16)h0[(size_t)b * h0_stride + hh];
  }

  // prefetch step 0 into buffer 0
  async_fetch_step(xz, xr, &xz_lds[0][0], &xr_lds[0][0], tid, 0);
  wait_async0();
  __syncthreads();

  for (int t = 0; t < Sdim; ++t) {
    const int cur = t & 1;
    if (t + 1 < Sdim)
      async_fetch_step(xz, xr, &xz_lds[cur ^ 1][0], &xr_lds[cur ^ 1][0], tid, t + 1);

    // --- z and r gate GEMMs over K = 512 ---
    v8f accz = {};
    v8f accr = {};
    for (int k = 0; k < Hdim; k += 32) {
      v16bf a  = load_a_frag(&h_bf[0][k], Hdim, lane);
      v16bf bz = load_b_frag(Whz, Hdim, n0, k, lane);
      v16bf br = load_b_frag(Whr, Hdim, n0, k, lane);
      accz = wmma_bf16f32(a, bz, accz);
      accr = wmma_bf16f32(a, br, accr);
    }

    float zv[8];
#pragma unroll
    for (int i = 0; i < 8; ++i) {
      int   m = mbase + i;
      float z = 1.0f / (1.0f + __expf(-(xz_lds[cur][m * Hdim + col] + accz[i])));
      float r = 1.0f / (1.0f + __expf(-(xr_lds[cur][m * Hdim + col] + accr[i])));
      zv[i] = z;
      rh_bf[m][col] = (bf16)(r * hreg[i]);
    }
    __syncthreads();  // rh visible to all waves

    // --- candidate gate GEMM: (r*h) @ Whr^T ---
    v8f accg = {};
    for (int k = 0; k < Hdim; k += 32) {
      v16bf a = load_a_frag(&rh_bf[0][k], Hdim, lane);
      v16bf b = load_b_frag(Whr, Hdim, n0, k, lane);
      accg = wmma_bf16f32(a, b, accg);
    }

    bf16* y_t = y + (size_t)t * (Bdim * Hdim);
#pragma unroll
    for (int i = 0; i < 8; ++i) {
      int   m = mbase + i;
      float e = __expf(2.0f * (xr_lds[cur][m * Hdim + col] + accg[i]));
      float g = (e - 1.0f) / (e + 1.0f);  // tanh
      float hn = zv[i] * hreg[i] + (1.0f - zv[i]) * g;
      hreg[i] = hn;
      y_t[(size_t)m * Hdim + col] = (bf16)hn;
      h_bf[m][col] = (bf16)hn;
    }
    wait_async0();    // next step's xz/xr slice fully in LDS
    __syncthreads();  // h_bf update + async data visible to all waves
  }

#pragma unroll
  for (int i = 0; i < 8; ++i)
    hT[(size_t)(mbase + i) * hT_stride + col] = hreg[i];
}

// ---------------------------------------------------------------------------
// Output head: out[b][t][o] = by[o] + sum_h y1[t][b][h] * Why[o][h]
// grid = (S, O/16), one wave per block.
// ---------------------------------------------------------------------------
__global__ void head_kernel(const bf16* __restrict__ Ybf, const bf16* __restrict__ Why,
                            const float* __restrict__ by, float* __restrict__ out) {
  const int t    = blockIdx.x;
  const int n0   = blockIdx.y * 16;
  const int lane = threadIdx.x & 31;

  const bf16* a_base = Ybf + (size_t)t * (Bdim * Hdim);
  v8f acc = {};
  for (int k = 0; k < Hdim; k += 32) {
    v16bf a = load_a_frag(a_base + k, Hdim, lane);
    v16bf b = load_b_frag(Why, Hdim, n0, k, lane);
    acc = wmma_bf16f32(a, b, acc);
  }

  const int   col   = n0 + (lane & 15);
  const int   mbase = (lane >> 4) << 3;
  const float bb    = by[col];
#pragma unroll
  for (int i = 0; i < 8; ++i) {
    int m = mbase + i;  // batch row
    out[(size_t)m * (Sdim * Odim) + (size_t)t * Odim + col] = acc[i] + bb;
  }
}

// ---------------------------------------------------------------------------
// Launch
// ---------------------------------------------------------------------------
extern "C" void kernel_launch(void* const* d_in, const int* in_sizes, int n_in,
                              void* d_out, int out_size, void* d_ws, size_t ws_size,
                              hipStream_t stream) {
  (void)in_sizes; (void)n_in; (void)out_size; (void)ws_size;

  const float* x    = (const float*)d_in[0];
  const float* hs   = (const float*)d_in[1];
  const float* Wxz0 = (const float*)d_in[2];
  const float* Whz0 = (const float*)d_in[3];
  const float* bhz0 = (const float*)d_in[4];
  const float* Wxr0 = (const float*)d_in[5];
  const float* Whr0 = (const float*)d_in[6];
  const float* bhr0 = (const float*)d_in[7];
  const float* Wxz1 = (const float*)d_in[8];
  const float* Whz1 = (const float*)d_in[9];
  const float* bhz1 = (const float*)d_in[10];
  const float* Wxr1 = (const float*)d_in[11];
  const float* Whr1 = (const float*)d_in[12];
  const float* bhr1 = (const float*)d_in[13];
  const float* Why  = (const float*)d_in[14];
  const float* by   = (const float*)d_in[15];
  float* out = (float*)d_out;

  // ---- carve workspace ----
  char*  ws  = (char*)d_ws;
  size_t off = 0;
  auto carve = [&](size_t bytes) -> char* {
    char* p = ws + off;
    off = (off + bytes + 255) & ~(size_t)255;
    return p;
  };
  bf16* wxz0b = (bf16*)carve((size_t)Hdim * Idim * sizeof(bf16));
  bf16* wxr0b = (bf16*)carve((size_t)Hdim * Idim * sizeof(bf16));
  bf16* whz0b = (bf16*)carve((size_t)Hdim * Hdim * sizeof(bf16));
  bf16* whr0b = (bf16*)carve((size_t)Hdim * Hdim * sizeof(bf16));
  bf16* wxz1b = (bf16*)carve((size_t)Hdim * Hdim * sizeof(bf16));
  bf16* wxr1b = (bf16*)carve((size_t)Hdim * Hdim * sizeof(bf16));
  bf16* whz1b = (bf16*)carve((size_t)Hdim * Hdim * sizeof(bf16));
  bf16* whr1b = (bf16*)carve((size_t)Hdim * Hdim * sizeof(bf16));
  bf16* whyb  = (bf16*)carve((size_t)Odim * Hdim * sizeof(bf16));
  bf16* xbf   = (bf16*)carve((size_t)Bdim * Sdim * Idim * sizeof(bf16));
  bf16* y0b   = (bf16*)carve((size_t)Sdim * Bdim * Hdim * sizeof(bf16));
  bf16* y1b   = (bf16*)carve((size_t)Sdim * Bdim * Hdim * sizeof(bf16));
  float* xzbuf = (float*)carve((size_t)Sdim * Bdim * Hdim * sizeof(float));
  float* xrbuf = (float*)carve((size_t)Sdim * Bdim * Hdim * sizeof(float));

  auto cvt = [&](const float* src, bf16* dst, int n) {
    cvt_f32_bf16<<<(n + 255) / 256, 256, 0, stream>>>(src, dst, n);
  };
  cvt(Wxz0, wxz0b, Hdim * Idim);
  cvt(Wxr0, wxr0b, Hdim * Idim);
  cvt(Whz0, whz0b, Hdim * Hdim);
  cvt(Whr0, whr0b, Hdim * Hdim);
  cvt(Wxz1, wxz1b, Hdim * Hdim);
  cvt(Wxr1, wxr1b, Hdim * Hdim);
  cvt(Whz1, whz1b, Hdim * Hdim);
  cvt(Whr1, whr1b, Hdim * Hdim);
  cvt(Why,  whyb,  Odim * Hdim);
  cvt(x,    xbf,   Bdim * Sdim * Idim);

  const size_t OUT_ELEMS = (size_t)Bdim * Sdim * Odim;
  dim3 gp(Sdim, Hdim / 16);

  // ---- layer 0 ----
  // x layout [B][S][I]: element (b,t,k) at b*S*I + t*I + k
  xproj_kernel<<<gp, 64, 0, stream>>>(xbf, (long)Idim, (long)((size_t)Sdim * Idim), Idim,
                                      wxz0b, wxr0b, bhz0, bhr0, xzbuf, xrbuf);
  gru_scan_kernel<<<1, 1024, 0, stream>>>(xzbuf, xrbuf, whz0b, whr0b,
                                          hs + 0 * Hdim, Ldim * Hdim,
                                          y0b, out + OUT_ELEMS + 0 * Hdim, Ldim * Hdim);

  // ---- layer 1 ----
  // y0 layout [S][B][H]: element (b,t,k) at t*B*H + b*H + k
  xproj_kernel<<<gp, 64, 0, stream>>>(y0b, (long)(Bdim * Hdim), (long)Hdim, Hdim,
                                      wxz1b, wxr1b, bhz1, bhr1, xzbuf, xrbuf);
  gru_scan_kernel<<<1, 1024, 0, stream>>>(xzbuf, xrbuf, whz1b, whr1b,
                                          hs + 1 * Hdim, Ldim * Hdim,
                                          y1b, out + OUT_ELEMS + 1 * Hdim, Ldim * Hdim);

  // ---- output head ----
  dim3 gh(Sdim, Odim / 16);
  head_kernel<<<gh, 32, 0, stream>>>(y1b, whyb, by, out);
}
